// BiMatchLSTM_49658411877023
// MI455X (gfx1250) — matrix operand order repaired and checked by reading, not compiled, not run on hardware
//
#include <hip/hip_runtime.h>
#include <hip/hip_bf16.h>
#include <math.h>

// ---------------- problem constants ----------------
#define B_   128
#define TP_  400
#define TQ_  60
#define DP_  300
#define DQ_  300
#define H_   150
#define XK   768    // padded K for gates GEMM: [p(300) | z(300) | h(150) | pad(18)]
#define NQ   608    // padded N for gates (4H = 600)
#define KP   320    // padded K for W_p (DP=300)
#define NR   160    // padded N for Gp/Gr (H=150), also padded K for W_r
#define NTILES_G (NQ / 16)   // 38

typedef __attribute__((ext_vector_type(16))) __bf16 v16bf;
typedef __attribute__((ext_vector_type(8)))  float  v8f;

// ---------------- fast math helpers ----------------
__device__ __forceinline__ float fast_tanh(float x) {
#if __has_builtin(__builtin_amdgcn_tanhf)
  return __builtin_amdgcn_tanhf(x);
#else
  return tanhf(x);
#endif
}
__device__ __forceinline__ float fast_sigmoid(float x) {
  return 1.0f / (1.0f + __expf(-x));
}

// ---------------- WMMA fragment loaders (CDNA5 wave32 layouts) ----------------
// A (16x32 bf16, MxK): lanes 0-15 -> M=lane, K = 0..15; lanes 16-31 -> K = 16..31
__device__ __forceinline__ v16bf load_a_frag(const __bf16* base, int ldk, int kBase, int lane) {
  const int m  = lane & 15;
  const int kh = (lane & 16) ? 16 : 0;
  const __bf16* row = base + (size_t)m * ldk + kBase;
  v16bf a;
#pragma unroll
  for (int j = 0; j < 4; ++j) {
    a[2 * j + 0] = row[kh + 2 * j + 0];
    a[2 * j + 1] = row[kh + 2 * j + 1];
    a[8 + 2 * j + 0] = row[8 + kh + 2 * j + 0];
    a[8 + 2 * j + 1] = row[8 + kh + 2 * j + 1];
  }
  return a;
}
// B (32x16 bf16, KxN), weight stored row-major [N][ldk] so B[k][n] = W[n*ldk + k].
// lanes 0-15 -> N=lane, K = 0..15; lanes 16-31 -> K = 16..31
__device__ __forceinline__ v16bf load_b_frag(const __bf16* W, int ldk, int nBase, int kBase, int lane) {
  const int n  = lane & 15;
  const int kh = (lane & 16) ? 16 : 0;
  const __bf16* col = W + (size_t)(nBase + n) * ldk + kBase + kh;
  v16bf b;
#pragma unroll
  for (int j = 0; j < 8; ++j) {
    b[2 * j + 0] = col[2 * j + 0];
    b[2 * j + 1] = col[2 * j + 1];
  }
  return b;
}

// ---------------- one-time prep: pack weights to bf16, fused biases ----------------
__global__ void __launch_bounds__(256) pack_kernel(
    const float* __restrict__ Wp,   const float* __restrict__ Wr,
    const float* __restrict__ WihF, const float* __restrict__ WhhF,
    const float* __restrict__ bihF, const float* __restrict__ bhhF,
    const float* __restrict__ WihB, const float* __restrict__ WhhB,
    const float* __restrict__ bihB, const float* __restrict__ bhhB,
    __bf16* __restrict__ WcatF, __bf16* __restrict__ WcatB,
    __bf16* __restrict__ WpP,   __bf16* __restrict__ WrP,
    float* __restrict__ biasF,  float* __restrict__ biasB)
{
  const int NCAT = NQ * XK;
  const int total = 2 * NCAT + NR * KP + NR * NR + 2 * 600;
  int idx = blockIdx.x * 256 + threadIdx.x;
  if (idx >= total) return;
  if (idx < 2 * NCAT) {
    int d = idx / NCAT, e = idx % NCAT;
    int n = e / XK, k = e % XK;
    const float* Wih = d ? WihB : WihF;
    const float* Whh = d ? WhhB : WhhF;
    float v = 0.0f;
    if (n < 600) {
      if (k < 600)      v = Wih[n * 600 + k];          // [p | z] input dims
      else if (k < 750) v = Whh[n * 150 + (k - 600)];  // h dims
    }
    (d ? WcatB : WcatF)[e] = (__bf16)v;
  } else if (idx < 2 * NCAT + NR * KP) {
    int e = idx - 2 * NCAT;
    int n = e / KP, k = e % KP;
    float v = (n < H_ && k < DP_) ? Wp[n * DP_ + k] : 0.0f;
    WpP[e] = (__bf16)v;
  } else if (idx < 2 * NCAT + NR * KP + NR * NR) {
    int e = idx - 2 * NCAT - NR * KP;
    int n = e / NR, k = e % NR;
    float v = (n < H_ && k < H_) ? Wr[n * H_ + k] : 0.0f;
    WrP[e] = (__bf16)v;
  } else {
    int e = idx - 2 * NCAT - NR * KP - NR * NR;
    int d = e / 600, n = e % 600;
    float v = d ? (bihB[n] + bhhB[n]) : (bihF[n] + bhhF[n]);
    (d ? biasB : biasF)[n] = v;
  }
}

// ---------------- one-time prep: Gq = einsum('btd,hd->bth', q, Wq) ----------------
__global__ void __launch_bounds__(256) gq_kernel(
    const float* __restrict__ q, const float* __restrict__ Wq, float* __restrict__ Gq)
{
  int idx = blockIdx.x * 256 + threadIdx.x;
  if (idx >= B_ * TQ_ * H_) return;
  int hh = idx % H_;
  int bt = idx / H_;
  const float* qr = q + (size_t)bt * DQ_;
  const float* wr = Wq + (size_t)hh * DQ_;
  float acc = 0.0f;
#pragma unroll 4
  for (int d = 0; d < DQ_; ++d) acc += qr[d] * wr[d];
  Gq[idx] = acc;
}

// ---------------- persistent Match-LSTM scan kernel ----------------
// grid.x = 16: blockIdx.x>>3 = direction (0 fwd, 1 bwd), blockIdx.x&7 = batch tile of 16 rows.
// 256 threads = 8 wave32s; dynamic LDS = 87,552 B.
#define SMEM_BYTES (16*XK*2 + 16*NR*4 + 16*152*4 + 16*64*4 + 16*NQ*4)

__global__ void __launch_bounds__(256) match_lstm_kernel(
    const float* __restrict__ input_p, const float* __restrict__ mask_p,
    const float* __restrict__ input_q, const float* __restrict__ mask_q,
    const float* __restrict__ Gq,
    const __bf16* __restrict__ WcatF, const __bf16* __restrict__ WcatB,
    const __bf16* __restrict__ WpP,   const __bf16* __restrict__ WrP,
    const float* __restrict__ biasF,  const float* __restrict__ biasB,
    const float* __restrict__ w_att,  const float* __restrict__ b_att,
    float* __restrict__ out)
{
  extern __shared__ char smemRaw[];
  __bf16* sX    = (__bf16*)smemRaw;                   // [16][XK]   x-concat (bf16 A matrix)
  float*  sGrGp = (float*)(smemRaw + 16 * XK * 2);    // [16][NR]   Gp + Gr
  float*  sH    = sGrGp + 16 * NR;                    // [16][152]  hidden state
  float*  sAl   = sH + 16 * 152;                      // [16][64]   attention scores / alpha
  float*  sGates= sAl + 16 * 64;                      // [16][NQ]   LSTM gates

  const int tid  = threadIdx.x;
  const int lane = tid & 31;
  const int wave = tid >> 5;
  const int dir  = blockIdx.x >> 3;
  const int b0   = (blockIdx.x & 7) * 16;

  const __bf16* Wcat = dir ? WcatB : WcatF;
  const float*  bias = dir ? biasB : biasF;
  const float   batt = b_att[0];

  // init: zero x-concat (incl. pads), zero h; c lives in registers (fixed thread->elem map)
  for (int i = tid; i < 16 * XK;  i += 256) sX[i] = (__bf16)0.0f;
  for (int i = tid; i < 16 * 152; i += 256) sH[i] = 0.0f;
  float cReg[10];
#pragma unroll
  for (int r = 0; r < 10; ++r) cReg[r] = 0.0f;
  __syncthreads();

  for (int s = 0; s < TP_; ++s) {
    const int t = dir ? (TP_ - 1 - s) : s;

    // ---- Phase A: stage p_t and h into x-concat (bf16) ----
    for (int i = tid; i < 16 * DP_; i += 256) {
      int bb = i / DP_, d = i - bb * DP_;
      sX[bb * XK + d] = (__bf16)input_p[((size_t)(b0 + bb) * TP_ + t) * DP_ + d];
    }
    for (int i = tid; i < 16 * H_; i += 256) {
      int bb = i / H_, hh = i - bb * H_;
      sX[bb * XK + 600 + hh] = (__bf16)sH[bb * 152 + hh];
    }
    __syncthreads();

    // ---- Phase B: GrGp = p_t @ Wp^T + h @ Wr^T (WMMA, M=16, N=160) ----
    for (int tc = 0; tc < 2; ++tc) {
      int tile = wave + tc * 8;
      if (tile < NR / 16) {
        v8f acc = {};
        for (int kb = 0; kb < KP; kb += 32) {     // Gp: A = p slice (zero-padded B nullifies k>=300)
          v16bf a  = load_a_frag(sX, XK, kb, lane);
          v16bf bm = load_b_frag(WpP, KP, tile * 16, kb, lane);
          acc = __builtin_amdgcn_wmma_f32_16x16x32_bf16(false, a, false, bm, (short)0, acc, false, false);
        }
        for (int kb = 0; kb < NR; kb += 32) {     // Gr: A = h slice at col 600
          v16bf a  = load_a_frag(sX, XK, 600 + kb, lane);
          v16bf bm = load_b_frag(WrP, NR, tile * 16, kb, lane);
          acc = __builtin_amdgcn_wmma_f32_16x16x32_bf16(false, a, false, bm, (short)0, acc, false, false);
        }
        int n = lane & 15, rowOff = (lane & 16) ? 8 : 0;
#pragma unroll
        for (int r = 0; r < 8; ++r) sGrGp[(rowOff + r) * NR + tile * 16 + n] = acc[r];
      }
    }
    __syncthreads();

    // ---- Phase C: attention scores alpha_raw[b][tq] = tanh(Gp+Gq+Gr) . w_att + b_att ----
    for (int p = tid; p < 16 * TQ_; p += 256) {
      int bb = p / TQ_, tq = p - bb * TQ_;
      const float* gq = Gq + ((size_t)(b0 + bb) * TQ_ + tq) * H_;
      const float* gr = sGrGp + bb * NR;
      float acc = batt;
#pragma unroll 2
      for (int hh = 0; hh < H_; ++hh)
        acc += fast_tanh(gq[hh] + gr[hh]) * w_att[hh];
      sAl[bb * 64 + tq] = acc;
    }
    __syncthreads();

    // ---- Phase D: masked softmax per batch row (reference semantics) ----
    if (tid < 16) {
      int bb = tid;
      const float* mq = mask_q + (size_t)(b0 + bb) * TQ_;
      float mx = -3.4e38f;
      for (int tq = 0; tq < TQ_; ++tq) {
        float x = sAl[bb * 64 + tq];
        x = fminf(fmaxf(x, -15.0f), 15.0f) * mq[tq];
        sAl[bb * 64 + tq] = x;
        mx = fmaxf(mx, x);
      }
      float sum = 0.0f;
      for (int tq = 0; tq < TQ_; ++tq) {
        float e = __expf(sAl[bb * 64 + tq] - mx) * mq[tq];
        sAl[bb * 64 + tq] = e;
        sum += e;
      }
      float inv = 1.0f / (sum + 1e-6f);
      for (int tq = 0; tq < TQ_; ++tq) sAl[bb * 64 + tq] *= inv;
    }
    __syncthreads();

    // ---- Phase E: z[b][d] = sum_tq alpha * input_q, write bf16 into x-concat ----
    for (int o = tid; o < 16 * DQ_; o += 256) {
      int bb = o / DQ_, d = o - bb * DQ_;
      const float* qb = input_q + (size_t)(b0 + bb) * TQ_ * DQ_ + d;
      float acc = 0.0f;
#pragma unroll 4
      for (int tq = 0; tq < TQ_; ++tq) acc += sAl[bb * 64 + tq] * qb[(size_t)tq * DQ_];
      sX[bb * XK + 300 + d] = (__bf16)acc;
    }
    __syncthreads();

    // ---- Phase F: gates = [p,z,h] @ Wcat^T + bias (WMMA, M=16, N=608, K=768) ----
    for (int i = 0; i < 5; ++i) {
      int tile = wave * 5 + i;
      if (tile < NTILES_G) {
        v8f acc = {};
        for (int kb = 0; kb < XK; kb += 32) {
          if ((kb & 127) == 0)
            __builtin_prefetch(Wcat + (size_t)(tile * 16) * XK + kb + 128, 0, 1);
          v16bf a  = load_a_frag(sX, XK, kb, lane);
          v16bf bm = load_b_frag(Wcat, XK, tile * 16, kb, lane);
          acc = __builtin_amdgcn_wmma_f32_16x16x32_bf16(false, a, false, bm, (short)0, acc, false, false);
        }
        int n = lane & 15, rowOff = (lane & 16) ? 8 : 0;
        int nG = tile * 16 + n;
        float bn = (nG < 600) ? bias[nG] : 0.0f;
#pragma unroll
        for (int r = 0; r < 8; ++r) sGates[(rowOff + r) * NQ + nG] = acc[r] + bn;
      }
    }
    __syncthreads();

    // ---- Phase G: LSTM pointwise + masked state blend + output write ----
    {
      int ridx = 0;
      for (int o = tid; o < 16 * H_; o += 256, ++ridx) {
        int bb = o / H_, hh = o - bb * H_;
        float ig = sGates[bb * NQ + hh];
        float fg = sGates[bb * NQ + 150 + hh];
        float gg = sGates[bb * NQ + 300 + hh];
        float og = sGates[bb * NQ + 450 + hh];
        float cPrev = cReg[ridx];
        float hPrev = sH[bb * 152 + hh];
        float cN = fast_sigmoid(fg) * cPrev + fast_sigmoid(ig) * fast_tanh(gg);
        float hN = fast_sigmoid(og) * fast_tanh(cN);
        float m = mask_p[(size_t)(b0 + bb) * TP_ + t];
        hN = hN * m + hPrev * (1.0f - m);
        cN = cN * m + cPrev * (1.0f - m);
        cReg[ridx] = cN;
        sH[bb * 152 + hh] = hN;
        // concat_states (masked by mask_p per reference), fwd at cols [0,150), bwd at [150,300)
        out[((size_t)(b0 + bb) * TP_ + t) * 300 + dir * 150 + hh] = hN * m;
        if (s == TP_ - 1)  // concat_last (unmasked)
          out[(size_t)B_ * TP_ * 300 + (size_t)(b0 + bb) * 300 + dir * 150 + hh] = hN;
      }
    }
    __syncthreads();
  }
}

// ---------------- host launcher ----------------
extern "C" void kernel_launch(void* const* d_in, const int* in_sizes, int n_in,
                              void* d_out, int out_size, void* d_ws, size_t ws_size,
                              hipStream_t stream) {
  (void)in_sizes; (void)n_in; (void)out_size; (void)ws_size;
  const float* input_p = (const float*)d_in[0];
  const float* mask_p  = (const float*)d_in[1];
  const float* input_q = (const float*)d_in[2];
  const float* mask_q  = (const float*)d_in[3];
  const float* W_p     = (const float*)d_in[4];
  const float* W_q     = (const float*)d_in[5];
  const float* W_r     = (const float*)d_in[6];
  const float* w_att   = (const float*)d_in[7];
  const float* b_att   = (const float*)d_in[8];
  const float* W_ih_f  = (const float*)d_in[9];
  const float* W_hh_f  = (const float*)d_in[10];
  const float* b_ih_f  = (const float*)d_in[11];
  const float* b_hh_f  = (const float*)d_in[12];
  const float* W_ih_b  = (const float*)d_in[13];
  const float* W_hh_b  = (const float*)d_in[14];
  const float* b_ih_b  = (const float*)d_in[15];
  const float* b_hh_b  = (const float*)d_in[16];

  // workspace carve (256B aligned)
  char* ws = (char*)d_ws;
  size_t off = 0;
  auto carve = [&](size_t bytes) { size_t o = off; off = (off + bytes + 255) & ~(size_t)255; return o; };
  float*  Gq    = (float*) (ws + carve((size_t)B_ * TQ_ * H_ * 4));
  __bf16* WcatF = (__bf16*)(ws + carve((size_t)NQ * XK * 2));
  __bf16* WcatB = (__bf16*)(ws + carve((size_t)NQ * XK * 2));
  __bf16* WpP   = (__bf16*)(ws + carve((size_t)NR * KP * 2));
  __bf16* WrP   = (__bf16*)(ws + carve((size_t)NR * NR * 2));
  float*  biasF = (float*) (ws + carve(600 * 4));
  float*  biasB = (float*) (ws + carve(600 * 4));

  // 1) pack weights to bf16 (L2-resident across the scan), fuse biases
  {
    const int total = 2 * NQ * XK + NR * KP + NR * NR + 2 * 600;
    pack_kernel<<<(total + 255) / 256, 256, 0, stream>>>(
        W_p, W_r, W_ih_f, W_hh_f, b_ih_f, b_hh_f,
        W_ih_b, W_hh_b, b_ih_b, b_hh_b,
        WcatF, WcatB, WpP, WrP, biasF, biasB);
  }
  // 2) hoist Gq out of the scan
  {
    const int total = B_ * TQ_ * H_;
    gq_kernel<<<(total + 255) / 256, 256, 0, stream>>>(input_q, W_q, Gq);
  }
  // 3) persistent bidirectional scan: 8 batch tiles x 2 directions
  match_lstm_kernel<<<dim3(16), dim3(256), SMEM_BYTES, stream>>>(
      input_p, mask_p, input_q, mask_q, Gq,
      WcatF, WcatB, WpP, WrP, biasF, biasB,
      w_att, b_att, (float*)d_out);
}
